// GraphConvClassifier_15590731284804
// MI455X (gfx1250) — matrix operand order repaired and checked by reading, not compiled, not run on hardware
//
#include <hip/hip_runtime.h>
#include <hip/hip_bf16.h>

// GraphConv x3 + global_max_pool + MLP for MI455X (gfx1250, wave32, WMMA).
//
// Per layer:
//   1) zero agg
//   2) edge scatter: agg[dst] += x[src] * w   (float4 gather + global_atomic_add_f32)
//   3) x' = relu(agg @ Wrel + brel + x @ Wroot)
//      LDS-staged f16 tiles (coalesced b128 loads, ds_load_b128 fragment reads)
//      -> v_wmma_f32_16x16x32_f16, f32 accumulate.
// Then segment_max pool (int-bitpattern atomicMax, valid since x>=0 post-relu),
// h = relu(g @ Wl0 + bl0) via the same WMMA kernel, final 32->1 dot.

typedef __attribute__((ext_vector_type(16))) _Float16 v16h;
typedef __attribute__((ext_vector_type(8)))  _Float16 v8h;
typedef __attribute__((ext_vector_type(8)))  float    v8f;

#define N_NODES 100000
#define N_EDGES 1600000
#define N_GRAPHS 512

// ---------------------------------------------------------------------------
// Simple kernels
// ---------------------------------------------------------------------------

__global__ void fill_zero_kernel(float* __restrict__ p, long long n) {
    long long i = (long long)blockIdx.x * blockDim.x + threadIdx.x;
    const long long stride = (long long)gridDim.x * blockDim.x;
    for (; i < n; i += stride) p[i] = 0.0f;
}

// agg[dst[e]] += x[src[e]] * w[e], float4 chunks; chunks = d/4 = 1<<shift.
// For d=128 a full wave shares one edge -> coalesced 512B gather + row-local atomics.
__global__ void scatter_agg_kernel(const float* __restrict__ x,
                                   const int* __restrict__ src,
                                   const int* __restrict__ dst,
                                   const float* __restrict__ w,
                                   float* __restrict__ agg,
                                   int E, int d, int shift) {
    const int mask = (1 << shift) - 1;
    const long long total = (long long)E << shift;
    const long long stride = (long long)gridDim.x * blockDim.x;
    for (long long gid = (long long)blockIdx.x * blockDim.x + threadIdx.x;
         gid < total; gid += stride) {
        const int e = (int)(gid >> shift);
        const int c = (int)gid & mask;
        const int s = src[e];
        const int t = dst[e];
        const float ww = w[e];
        const float4 v = *(const float4*)(x + (size_t)s * d + (size_t)c * 4);
        float* o = agg + (size_t)t * d + (size_t)c * 4;
        unsafeAtomicAdd(o + 0, v.x * ww);   // global_atomic_add_f32
        unsafeAtomicAdd(o + 1, v.y * ww);
        unsafeAtomicAdd(o + 2, v.z * ww);
        unsafeAtomicAdd(o + 3, v.w * ww);
    }
}

// Segment max over non-negative floats (int bit patterns order-preserving).
// g pre-zeroed; d = 64 = 1<<6.
__global__ void pool_max_kernel(const float* __restrict__ x,
                                const int* __restrict__ batch,
                                float* __restrict__ g, int Nn) {
    const long long total = (long long)Nn << 6;
    const long long stride = (long long)gridDim.x * blockDim.x;
    for (long long gid = (long long)blockIdx.x * blockDim.x + threadIdx.x;
         gid < total; gid += stride) {
        const int n = (int)(gid >> 6);
        const int c = (int)gid & 63;
        const int b = batch[n];
        const float v = x[((size_t)n << 6) + c];
        atomicMax((int*)(g + ((size_t)b << 6) + c), __float_as_int(v));
    }
}

// out[g] = sum_j h[g][j] * W[j] + b
__global__ void final_linear_kernel(const float* __restrict__ h,
                                    const float* __restrict__ W,
                                    const float* __restrict__ b,
                                    float* __restrict__ out, int G, int K) {
    const int g = blockIdx.x * blockDim.x + threadIdx.x;
    if (g >= G) return;
    float s = b[0];
    for (int j = 0; j < K; ++j) s += h[(size_t)g * K + j] * W[j];
    out[g] = s;
}

// ---------------------------------------------------------------------------
// LDS-staged WMMA update: out = relu(agg @ Wrel + brel [+ xin @ Wroot])
//   - block = 256 threads = 8 waves; each wave owns a 16-row M-tile (128 rows/block)
//   - K staged in 32-wide f16 chunks in LDS; weights staged transposed [col][k]
//   - fragments read as 2x ds_load_b128 per operand
//   - NT = Nout/16 accumulator tiles held in VGPRs across the K loop
// Requires: K % 32 == 0, M % 16 == 0.
// ---------------------------------------------------------------------------

template <bool HAS_ROOT, int NT>
__global__ void __launch_bounds__(256)
wmma_update_kernel(const float* __restrict__ agg,
                   const float* __restrict__ xin,
                   const float* __restrict__ Wrel,
                   const float* __restrict__ brel,
                   const float* __restrict__ Wroot,
                   float* __restrict__ out,
                   int M, int K) {
    constexpr int Nout = NT * 16;
    __shared__ _Float16 sA[128 * 32];        // agg tile   [row][k]
    __shared__ _Float16 sX[128 * 32];        // x tile     [row][k]   (HAS_ROOT)
    __shared__ _Float16 sWr[Nout * 32];      // Wrel  chunk, transposed [col][k]
    __shared__ _Float16 sWo[Nout * 32];      // Wroot chunk, transposed [col][k]

    const int t     = threadIdx.x;
    const int wave  = t >> 5;
    const int lane  = t & 31;
    const int half  = (lane >> 4) & 1;
    const int mrow  = lane & 15;             // row within wave's 16-row tile
    const int mbase = blockIdx.x * 128;      // first row of block's tile
    const int m0    = mbase + wave * 16;     // first row of wave's tile

    v8f acc[NT];
#pragma unroll
    for (int nt = 0; nt < NT; ++nt) acc[nt] = (v8f){};

    for (int k0 = 0; k0 < K; k0 += 32) {
        __syncthreads();

        // ---- stage A (and X) tiles: 128 rows x 32 floats -> f16 -------------
        // 1024 float4 chunks, 256 threads x 4
#pragma unroll
        for (int j = 0; j < 4; ++j) {
            const int f   = t + j * 256;     // [0,1024)
            const int row = f >> 3;          // 8 float4 per 32-float row
            const int cp  = (f & 7) * 4;
            const int grow = mbase + row;
            float4 va = (grow < M) ? *(const float4*)(agg + (size_t)grow * K + k0 + cp)
                                   : make_float4(0.f, 0.f, 0.f, 0.f);
            _Float16* pa = &sA[row * 32 + cp];
            pa[0] = (_Float16)va.x; pa[1] = (_Float16)va.y;
            pa[2] = (_Float16)va.z; pa[3] = (_Float16)va.w;
            if (HAS_ROOT) {
                float4 vx = (grow < M) ? *(const float4*)(xin + (size_t)grow * K + k0 + cp)
                                       : make_float4(0.f, 0.f, 0.f, 0.f);
                _Float16* px = &sX[row * 32 + cp];
                px[0] = (_Float16)vx.x; px[1] = (_Float16)vx.y;
                px[2] = (_Float16)vx.z; px[3] = (_Float16)vx.w;
            }
        }

        // ---- stage weights transposed: W[k0+kk][col] -> sW[col][kk] ---------
        // Nout*32 halves; Nout/64 float4-chunks... generic: (Nout*32/4) float4
        {
            const int nf4 = Nout * 8;        // float4 chunks (Nout*32/4)
            for (int f = t; f < nf4; f += 256) {
                const int kk = f >> (NT == 4 ? 4 : 3);          // f / (Nout/4)
                const int c  = (f & ((Nout >> 2) - 1)) * 4;
                const float4 wr = *(const float4*)(Wrel + (size_t)(k0 + kk) * Nout + c);
                sWr[(c + 0) * 32 + kk] = (_Float16)wr.x;
                sWr[(c + 1) * 32 + kk] = (_Float16)wr.y;
                sWr[(c + 2) * 32 + kk] = (_Float16)wr.z;
                sWr[(c + 3) * 32 + kk] = (_Float16)wr.w;
                if (HAS_ROOT) {
                    const float4 wo = *(const float4*)(Wroot + (size_t)(k0 + kk) * Nout + c);
                    sWo[(c + 0) * 32 + kk] = (_Float16)wo.x;
                    sWo[(c + 1) * 32 + kk] = (_Float16)wo.y;
                    sWo[(c + 2) * 32 + kk] = (_Float16)wo.z;
                    sWo[(c + 3) * 32 + kk] = (_Float16)wo.w;
                }
            }
        }

        __syncthreads();

        // ---- A fragments: 2x b128 from LDS ---------------------------------
        // element i<8 -> K = half*8 + i ; i>=8 -> K = 16 + half*8 + (i-8)
        const _Float16* arow = &sA[(wave * 16 + mrow) * 32];
        v8h alo = *(const v8h*)(arow + half * 8);
        v8h ahi = *(const v8h*)(arow + 16 + half * 8);
        v16h afrag = __builtin_shufflevector(alo, ahi, 0, 1, 2, 3, 4, 5, 6, 7,
                                             8, 9, 10, 11, 12, 13, 14, 15);
        v16h xfrag;
        if (HAS_ROOT) {
            const _Float16* xrow = &sX[(wave * 16 + mrow) * 32];
            v8h xlo = *(const v8h*)(xrow + half * 8);
            v8h xhi = *(const v8h*)(xrow + 16 + half * 8);
            xfrag = __builtin_shufflevector(xlo, xhi, 0, 1, 2, 3, 4, 5, 6, 7,
                                            8, 9, 10, 11, 12, 13, 14, 15);
        }

        // ---- per n-tile: B fragments (2x b128) + WMMA ----------------------
#pragma unroll
        for (int nt = 0; nt < NT; ++nt) {
            const int col = nt * 16 + (lane & 15);
            // element i<8 -> K = half*8 + i ; i>=8 -> K = 16 + half*8 + (i-8)
            const _Float16* bcol = &sWr[col * 32 + half * 8];
            v8h blo = *(const v8h*)(bcol);
            v8h bhi = *(const v8h*)(bcol + 16);
            v16h bfrag = __builtin_shufflevector(blo, bhi, 0, 1, 2, 3, 4, 5, 6, 7,
                                                 8, 9, 10, 11, 12, 13, 14, 15);
            acc[nt] = __builtin_amdgcn_wmma_f32_16x16x32_f16(
                false, afrag, false, bfrag, (short)0, acc[nt], false, false);
            if (HAS_ROOT) {
                const _Float16* ocol = &sWo[col * 32 + half * 8];
                v8h olo = *(const v8h*)(ocol);
                v8h ohi = *(const v8h*)(ocol + 16);
                v16h ofrag = __builtin_shufflevector(olo, ohi, 0, 1, 2, 3, 4, 5, 6, 7,
                                                     8, 9, 10, 11, 12, 13, 14, 15);
                acc[nt] = __builtin_amdgcn_wmma_f32_16x16x32_f16(
                    false, xfrag, false, ofrag, (short)0, acc[nt], false, false);
            }
        }
    }

    // ---- bias + relu + store -----------------------------------------------
    // C/D layout: VGPR r -> row m0 + r + 8*half, col nt*16 + (lane&15)
    if (m0 < M) {                            // M % 16 == 0 -> whole tile valid
        const int rbase = half * 8;
#pragma unroll
        for (int nt = 0; nt < NT; ++nt) {
            const int col = nt * 16 + (lane & 15);
            const float bias = brel[col];
#pragma unroll
            for (int r = 0; r < 8; ++r) {
                float v = acc[nt][r] + bias;
                v = v > 0.0f ? v : 0.0f;
                out[(size_t)(m0 + rbase + r) * Nout + col] = v;
            }
        }
    }
}

// ---------------------------------------------------------------------------
// Launch
// ---------------------------------------------------------------------------

extern "C" void kernel_launch(void* const* d_in, const int* in_sizes, int n_in,
                              void* d_out, int out_size, void* d_ws, size_t ws_size,
                              hipStream_t stream) {
    const float* x0   = (const float*)d_in[0];   // [N_NODES, 128]
    const int*   eidx = (const int*)  d_in[1];   // [2, N_EDGES]
    const int*   batch= (const int*)  d_in[2];   // [N_NODES]
    const float* ew   = (const float*)d_in[3];   // [N_EDGES]
    const float* Wrel[3] = { (const float*)d_in[4], (const float*)d_in[7],  (const float*)d_in[10] };
    const float* brel[3] = { (const float*)d_in[5], (const float*)d_in[8],  (const float*)d_in[11] };
    const float* Wroot[3]= { (const float*)d_in[6], (const float*)d_in[9],  (const float*)d_in[12] };
    const float* Wl0 = (const float*)d_in[13];
    const float* bl0 = (const float*)d_in[14];
    const float* Wl1 = (const float*)d_in[15];
    const float* bl1 = (const float*)d_in[16];
    float* out = (float*)d_out;

    const int* src = eidx;             // edge_index[0]
    const int* dst = eidx + N_EDGES;   // edge_index[1]

    // Workspace carve-up (~103 MB)
    char* ws = (char*)d_ws;
    float* agg = (float*)(ws);                                   // 100000*128*4 = 51.2 MB
    float* xb1 = (float*)(ws + (size_t)51200000);                // 100000*64*4  = 25.6 MB
    float* xb2 = (float*)(ws + (size_t)51200000 + 25600000);     // 25.6 MB
    float* gpool = (float*)(ws + (size_t)51200000 + 2*25600000); // 512*64*4
    float* hbuf  = (float*)((char*)gpool + (size_t)N_GRAPHS * 64 * 4);

    const int dims[3] = { 128, 64, 64 };       // feature dim entering layer i
    const float* xin = x0;
    float* xbufs[3] = { xb1, xb2, xb1 };

    for (int layer = 0; layer < 3; ++layer) {
        const int K = dims[layer];
        const int shift = (K == 128) ? 5 : 4;  // log2(K/4)
        const long long aggN = (long long)N_NODES * K;

        fill_zero_kernel<<<2048, 256, 0, stream>>>(agg, aggN);

        scatter_agg_kernel<<<4096, 256, 0, stream>>>(xin, src, dst, ew, agg,
                                                     N_EDGES, K, shift);

        const int blocks = (N_NODES + 127) / 128;    // 782 (last block partial)
        wmma_update_kernel<true, 4><<<blocks, 256, 0, stream>>>(
            agg, xin, Wrel[layer], brel[layer], Wroot[layer],
            xbufs[layer], N_NODES, K);

        xin = xbufs[layer];
    }

    // Global max pool: g[512,64]
    fill_zero_kernel<<<64, 256, 0, stream>>>(gpool, (long long)N_GRAPHS * 64);
    pool_max_kernel<<<4096, 256, 0, stream>>>(xin, batch, gpool, N_NODES);

    // h = relu(g @ Wl0 + bl0): M=512, K=64, Nout=32 (NT=2), no root term
    wmma_update_kernel<false, 2><<<512 / 128, 256, 0, stream>>>(
        gpool, nullptr, Wl0, bl0, nullptr, hbuf, N_GRAPHS, 64);

    // out = h @ Wl1 + bl1
    final_linear_kernel<<<2, 256, 0, stream>>>(hbuf, Wl1, bl1, out, N_GRAPHS, 32);
}
